// CMEN_90091234001030
// MI455X (gfx1250) — matrix operand order repaired and checked by reading, not compiled, not run on hardware
//
#include <hip/hip_runtime.h>
#include <cstdint>

typedef _Float16 half_t;
typedef half_t v16h __attribute__((ext_vector_type(16)));
typedef half_t v8h  __attribute__((ext_vector_type(8)));
typedef float  v8f  __attribute__((ext_vector_type(8)));
typedef unsigned int u32x4 __attribute__((ext_vector_type(4)));
typedef int i32x8 __attribute__((ext_vector_type(8)));
typedef int i32x4 __attribute__((ext_vector_type(4)));

#define BB   256
#define KKY  512
#define DD   384
#define HH   128
#define BBKK (BB * KKY)

__device__ __forceinline__ float silu_f(float x) { return x / (1.f + __expf(-x)); }

// ---------------------------------------------------------------------------
// Tensor Data Mover: load a [16 x tileW] tile (row stride = lda elems) from
// global memory into LDS (compacted row-major). D# built per ISA 08 §8.
// dszCode: 1 = 2-byte elems, 2 = 4-byte elems.
// Toolchain here is the 6-arg builtin:
//   (uint32x4 g0, int32x8 g1, int32x4 g2, int32x4 g3, int32x8 extra, i32 cpol)
// ---------------------------------------------------------------------------
__device__ __forceinline__ void tdm_load_2d(const void* gsrc, void* ldsDst,
                                            int tileWElems, int ldaElems, int dszCode,
                                            int lane) {
#if __has_builtin(__builtin_amdgcn_tensor_load_to_lds)
  uint64_t ga  = (uint64_t)(uintptr_t)gsrc;
  uint32_t glo = (uint32_t)__builtin_amdgcn_readfirstlane((int)(uint32_t)ga);
  uint32_t ghi = (uint32_t)__builtin_amdgcn_readfirstlane((int)(uint32_t)(ga >> 32));
  uint32_t lds = (uint32_t)__builtin_amdgcn_readfirstlane((int)(uint32_t)(uintptr_t)ldsDst);
  u32x4 g0;
  g0[0] = 1u;                                    // count=1, user mode, no gather
  g0[1] = lds;                                   // lds_addr
  g0[2] = glo;                                   // global_addr[31:0]
  g0[3] = (ghi & 0x01FFFFFFu) | (2u << 30);      // global_addr[56:32] | type=2
  i32x8 g1;
  uint32_t td0 = (uint32_t)ldaElems, td1 = 16u;
  g1[0] = (int)((uint32_t)dszCode << 16);        // wg_mask=0, data_size
  g1[1] = (int)((td0 & 0xFFFFu) << 16);          // tensor_dim0[15:0] (bits 63:48)
  g1[2] = (int)(((td0 >> 16) & 0xFFFFu) | ((td1 & 0xFFFFu) << 16));
  g1[3] = (int)(((td1 >> 16) & 0xFFFFu) | (((uint32_t)tileWElems & 0xFFFFu) << 16)); // tile_dim0
  g1[4] = (int)16u;                              // tile_dim1=16, tile_dim2=0
  uint64_t s0 = (uint64_t)(uint32_t)ldaElems;    // tensor_dim0_stride
  g1[5] = (int)(uint32_t)(s0 & 0xFFFFFFFFu);
  g1[6] = (int)(uint32_t)((s0 >> 32) & 0xFFFFu);
  g1[7] = 0;
  i32x4 gz4 = {0, 0, 0, 0};
  i32x8 gz8 = {0, 0, 0, 0, 0, 0, 0, 0};
  __builtin_amdgcn_tensor_load_to_lds(g0, g1, gz4, gz4, gz8, 0);
#else
  // cooperative fallback (no TDM builtin)
  if (dszCode == 2) {
    const float* s = (const float*)gsrc; float* d = (float*)ldsDst;
    for (int r = 0; r < 16; ++r)
      for (int c = lane; c < tileWElems; c += 32)
        d[r * tileWElems + c] = s[(size_t)r * ldaElems + c];
  } else {
    const half_t* s = (const half_t*)gsrc; half_t* d = (half_t*)ldsDst;
    for (int r = 0; r < 16; ++r)
      for (int c = lane; c < tileWElems; c += 32)
        d[r * tileWElems + c] = s[(size_t)r * ldaElems + c];
  }
#endif
}

// ---------------------------------------------------------------------------
// Pack f32 weight W[N, Kd] (row-major, out x in) into WMMA B-operand layout:
// Bp[((nt*KSTEPS + kt)*32 + lane)*16 + i] = (f16) W[n, k]
//   n = nt*16 + (lane & 15);  k = kt*32 + (lane<16 ? 0 : 8) + (i<8 ? i : i+8)
// Each lane then fetches its whole v16h fragment with one 32B load.
// ---------------------------------------------------------------------------
__global__ void pack_b_kernel(const float* __restrict__ W, int N, int Kd,
                              half_t* __restrict__ Bp) {
  int idx = blockIdx.x * blockDim.x + threadIdx.x;
  int ksteps = Kd >> 5;
  int total = (N >> 4) * ksteps * 512;
  if (idx >= total) return;
  int i    = idx & 15;
  int lane = (idx >> 4) & 31;
  int t    = idx >> 9;
  int kt   = t % ksteps;
  int nt   = t / ksteps;
  int n = nt * 16 + (lane & 15);
  int k = kt * 32 + ((lane < 16) ? 0 : 8) + ((i < 8) ? i : i + 8);
  Bp[idx] = (half_t)W[(size_t)n * Kd + k];
}

// ---------------------------------------------------------------------------
// WMMA GEMM: C[Mrows, N](f16) = epi( A[Mrows, Kd] @ packedB + bias )
// One wave per block; each block computes a 16-row stripe across all N.
// A tile staged in LDS via TDM; AF32 converts f32->f16 on the fly.
// EPI: 0 = identity, 1 = SiLU.
// ---------------------------------------------------------------------------
template <bool AF32, int KD, int EPI>
__global__ __launch_bounds__(32) void wmma_gemm_kernel(
    const void* __restrict__ Av, int lda, int Mrows,
    const half_t* __restrict__ Bp, const float* __restrict__ bias,
    half_t* __restrict__ C, int N) {
  constexpr int ESZ = AF32 ? 4 : 2;
  constexpr int KSTEPS = KD / 32;
  __shared__ __align__(16) char smRaw[16 * KD * ESZ];
  const int lane = threadIdx.x;
  const int rowTile = blockIdx.x;
  if (rowTile * 16 >= Mrows) return;

  const char* gsrc = (const char*)Av + (size_t)rowTile * 16 * lda * ESZ;
  tdm_load_2d(gsrc, smRaw, KD, lda, AF32 ? 2 : 1, lane);
  __builtin_amdgcn_s_wait_tensorcnt(0);
  __syncthreads();

  const int r  = lane & 15;
  const int kb = (lane < 16) ? 0 : 8;   // A-matrix 16-bit lane layout
  const int ntiles = N >> 4;

  for (int nt = 0; nt < ntiles; ++nt) {
    union { v8f v; float f[8]; } cu;
#pragma unroll
    for (int z = 0; z < 8; ++z) cu.f[z] = 0.f;

    for (int kt = 0; kt < KSTEPS; ++kt) {
      union { v16h v; half_t h[16]; } au;
      if (AF32) {
        const float* row = (const float*)smRaw + r * KD + kt * 32 + kb;
        float4 f0 = *(const float4*)(row);
        float4 f1 = *(const float4*)(row + 4);
        float4 f2 = *(const float4*)(row + 16);
        float4 f3 = *(const float4*)(row + 20);
        au.h[0] = (half_t)f0.x;  au.h[1] = (half_t)f0.y;
        au.h[2] = (half_t)f0.z;  au.h[3] = (half_t)f0.w;
        au.h[4] = (half_t)f1.x;  au.h[5] = (half_t)f1.y;
        au.h[6] = (half_t)f1.z;  au.h[7] = (half_t)f1.w;
        au.h[8]  = (half_t)f2.x; au.h[9]  = (half_t)f2.y;
        au.h[10] = (half_t)f2.z; au.h[11] = (half_t)f2.w;
        au.h[12] = (half_t)f3.x; au.h[13] = (half_t)f3.y;
        au.h[14] = (half_t)f3.z; au.h[15] = (half_t)f3.w;
      } else {
        const half_t* row = (const half_t*)smRaw + r * KD + kt * 32 + kb;
        v8h lo = *(const v8h*)(row);
        v8h hi = *(const v8h*)(row + 16);
#pragma unroll
        for (int z = 0; z < 8; ++z) { au.h[z] = lo[z]; au.h[8 + z] = hi[z]; }
      }
      const half_t* bptr = Bp + ((((size_t)nt * KSTEPS) + kt) * 32 + lane) * 16;
      v16h bfrag = *(const v16h*)bptr;
      __builtin_prefetch(bptr + 512, 0, 1);  // global_prefetch next fragment block
      cu.v = __builtin_amdgcn_wmma_f32_16x16x32_f16(
          false, au.v, false, bfrag, (short)0, cu.v, false, false);
    }

    const int col = nt * 16 + (lane & 15);
    const int rbase = rowTile * 16 + ((lane < 16) ? 0 : 8);
    const float bv = bias ? bias[col] : 0.f;
#pragma unroll
    for (int v2 = 0; v2 < 8; ++v2) {
      float x = cu.f[v2] + bv;
      if (EPI == 1) x = silu_f(x);
      C[(size_t)(rbase + v2) * N + col] = (half_t)x;
    }
  }
}

// ---------------------------------------------------------------------------
// Relevance attention: Q = qW@h_q+qb; scores over Kp (P cols 0..127);
// + log(clip(y)); softmax; ctx from V (P cols 128..255); writes [ctx, h_q]
// concat (f32) for the following h1 WMMA GEMM.
// ---------------------------------------------------------------------------
__global__ void rel_attn_kernel(const float* __restrict__ hq_g,
                                const float* __restrict__ qW,
                                const float* __restrict__ qb,
                                const half_t* __restrict__ P,
                                const float* __restrict__ y,
                                float* __restrict__ cc) {
  __shared__ float hq[DD];
  __shared__ float Qs[HH];
  __shared__ float sc[KKY];
  __shared__ float red[128];
  const int b = blockIdx.x, t = threadIdx.x;
  for (int d = t; d < DD; d += 128) hq[d] = hq_g[b * DD + d];
  __syncthreads();
  if (t < HH) {
    float s = qb[t];
    const float* w = qW + t * DD;
    for (int d = 0; d < DD; ++d) s += w[d] * hq[d];
    Qs[t] = s;
  }
  __syncthreads();
  for (int k = t; k < KKY; k += 128) {
    const half_t* kp = P + (size_t)(b * KKY + k) * 384;
    float s = 0.f;
    for (int j = 0; j < HH; ++j) s += Qs[j] * (float)kp[j];
    sc[k] = s * 0.08838834764831845f + logf(fmaxf(y[b * KKY + k], 1e-8f));
  }
  __syncthreads();
  float lm = -1e30f;
  for (int k = t; k < KKY; k += 128) lm = fmaxf(lm, sc[k]);
  red[t] = lm; __syncthreads();
  for (int s2 = 64; s2 > 0; s2 >>= 1) { if (t < s2) red[t] = fmaxf(red[t], red[t + s2]); __syncthreads(); }
  const float mx = red[0]; __syncthreads();
  float ls = 0.f;
  for (int k = t; k < KKY; k += 128) { float e = __expf(sc[k] - mx); sc[k] = e; ls += e; }
  red[t] = ls; __syncthreads();
  for (int s2 = 64; s2 > 0; s2 >>= 1) { if (t < s2) red[t] += red[t + s2]; __syncthreads(); }
  const float inv = 1.f / red[0]; __syncthreads();
  if (t < HH) {
    float c = 0.f;
    for (int k = 0; k < KKY; ++k) c += sc[k] * (float)P[(size_t)(b * KKY + k) * 384 + 128 + t];
    cc[b * 512 + t] = c * inv;
  }
  for (int d = t; d < DD; d += 128) cc[b * 512 + 128 + d] = hq[d];
}

// ---------------------------------------------------------------------------
// MHA attention for composition hops (4 heads x 32 dims, key_padding y<0.05)
// ---------------------------------------------------------------------------
__global__ void mha_attn_kernel(const half_t* __restrict__ q,
                                const half_t* __restrict__ kh,
                                const half_t* __restrict__ vh,
                                const float* __restrict__ y,
                                half_t* __restrict__ ctxc) {
  __shared__ float qs[HH];
  __shared__ float sc[KKY];
  __shared__ float red[128];
  const int b = blockIdx.x, t = threadIdx.x;
  if (t < HH) qs[t] = (float)q[b * HH + t];
  __syncthreads();
  for (int h = 0; h < 4; ++h) {
    for (int k = t; k < KKY; k += 128) {
      const half_t* kr = kh + (size_t)(b * KKY + k) * HH + h * 32;
      float s = 0.f;
      for (int d = 0; d < 32; ++d) s += qs[h * 32 + d] * (float)kr[d];
      s *= 0.17677669529663687f;
      sc[k] = (y[b * KKY + k] < 0.05f) ? -1e9f : s;
    }
    __syncthreads();
    float lm = -1e30f;
    for (int k = t; k < KKY; k += 128) lm = fmaxf(lm, sc[k]);
    red[t] = lm; __syncthreads();
    for (int s2 = 64; s2 > 0; s2 >>= 1) { if (t < s2) red[t] = fmaxf(red[t], red[t + s2]); __syncthreads(); }
    const float mx = red[0]; __syncthreads();
    float ls = 0.f;
    for (int k = t; k < KKY; k += 128) { float e = __expf(sc[k] - mx); sc[k] = e; ls += e; }
    red[t] = ls; __syncthreads();
    for (int s2 = 64; s2 > 0; s2 >>= 1) { if (t < s2) red[t] += red[t + s2]; __syncthreads(); }
    const float inv = 1.f / red[0]; __syncthreads();
    if ((t >> 5) == h) {
      float c = 0.f;
      for (int k = 0; k < KKY; ++k) c += sc[k] * (float)vh[(size_t)(b * KKY + k) * HH + t];
      ctxc[b * HH + t] = (half_t)(c * inv);
    }
    __syncthreads();
  }
}

// --------------------------- top-k (descending) ----------------------------
__global__ void topk_kernel(const float* __restrict__ y, int* __restrict__ idx, int kk) {
  __shared__ float ys[KKY];
  const int b = blockIdx.x, t = threadIdx.x;
  for (int i = t; i < KKY; i += 32) ys[i] = y[b * KKY + i];
  __syncthreads();
  if (t == 0) {
    for (int s = 0; s < kk; ++s) {
      int am = 0; float mv = -1e30f;
      for (int i = 0; i < KKY; ++i) if (ys[i] > mv) { mv = ys[i]; am = i; }
      idx[b * kk + s] = am; ys[am] = -1e30f;
    }
  }
}

// --------------------------- temporal pair MLP -----------------------------
__global__ void temporal_kernel(const float* __restrict__ M_, const float* __restrict__ y,
                                const float* __restrict__ ts, const int* __restrict__ idx10,
                                const float* __restrict__ w1, const float* __restrict__ b1,
                                const float* __restrict__ w2, const float* __restrict__ b2,
                                float* __restrict__ out) {
  __shared__ float mi[DD], mj[DD], hb[64];
  const int blk = blockIdx.x;
  const int b = blk / 45, p = blk % 45;
  int i = 0, rem = p, cnt = 9;
  while (rem >= cnt) { rem -= cnt; ++i; --cnt; }
  const int j = i + 1 + rem;
  const int gi = idx10[b * 10 + i], gj = idx10[b * 10 + j];
  const int t = threadIdx.x;  // 64
  const float* Mi = M_ + (size_t)(b * KKY + gi) * DD;
  const float* Mj = M_ + (size_t)(b * KKY + gj) * DD;
  for (int d = t; d < DD; d += 64) { mi[d] = Mi[d]; mj[d] = Mj[d]; }
  __syncthreads();
  const float dt = fabsf(ts[b * KKY + gi] - ts[b * KKY + gj]);
  const float yi = y[b * KKY + gi], yj = y[b * KKY + gj];
  const float* w = w1 + (size_t)t * 769;
  float acc = b1[t];
  for (int d = 0; d < DD; ++d) acc += w[d] * mi[d];
  for (int d = 0; d < DD; ++d) acc += w[DD + d] * mj[d];
  acc += w[768] * dt;
  hb[t] = silu_f(acc) * w2[t];
  __syncthreads();
  for (int s = 32; s > 0; s >>= 1) { if (t < s) hb[t] += hb[t + s]; __syncthreads(); }
  if (t == 0) atomicAdd(&out[b], (hb[0] + b2[0]) * yi * yj);
}

// --------------------------- recency pair MLP ------------------------------
__global__ void recency_kernel(const float* __restrict__ M_, const float* __restrict__ y,
                               const float* __restrict__ ts, const int* __restrict__ idx8,
                               const float* __restrict__ w1, const float* __restrict__ b1,
                               const float* __restrict__ w2, const float* __restrict__ b2,
                               const float* __restrict__ w3, const float* __restrict__ b3,
                               float* __restrict__ out) {
  __shared__ float mi[DD], mj[DD], h1s[64], h2s[32];
  const int blk = blockIdx.x;
  const int b = blk / 28, p = blk % 28;
  int i = 0, rem = p, cnt = 7;
  while (rem >= cnt) { rem -= cnt; ++i; --cnt; }
  const int j = i + 1 + rem;
  const int gi = idx8[b * 8 + i], gj = idx8[b * 8 + j];
  const int t = threadIdx.x;  // 64
  const float* Mi = M_ + (size_t)(b * KKY + gi) * DD;
  const float* Mj = M_ + (size_t)(b * KKY + gj) * DD;
  for (int d = t; d < DD; d += 64) { mi[d] = Mi[d]; mj[d] = Mj[d]; }
  __syncthreads();
  const float dt = ts[b * KKY + gi] - ts[b * KKY + gj];
  const float yi = y[b * KKY + gi], yj = y[b * KKY + gj];
  const float* w = w1 + (size_t)t * 771;
  float acc = b1[t];
  for (int d = 0; d < DD; ++d) acc += w[d] * mi[d];
  for (int d = 0; d < DD; ++d) acc += w[DD + d] * mj[d];
  acc += w[768] * dt + w[769] * yi + w[770] * yj;
  h1s[t] = silu_f(acc);
  __syncthreads();
  if (t < 32) {
    float a2 = b2[t];
    const float* ww = w2 + t * 64;
    for (int u = 0; u < 64; ++u) a2 += ww[u] * h1s[u];
    h2s[t] = silu_f(a2) * w3[t];
  }
  __syncthreads();
  if (t == 0) {
    float c = b3[0];
    for (int u = 0; u < 32; ++u) c += h2s[u];
    atomicAdd(&out[b], c * yi * yj);
  }
}

// --------------------------- small utility kernels -------------------------
__global__ void weighted_kernel(const float* __restrict__ M_, const float* __restrict__ y,
                                float* __restrict__ wsum) {
  __shared__ float ysh[KKY];
  const int b = blockIdx.x, t = threadIdx.x;  // 384 threads
  for (int i = t; i < KKY; i += 384) ysh[i] = y[b * KKY + i];
  __syncthreads();
  float s = 0.f;
  for (int k = 0; k < KKY; ++k) s += M_[(size_t)(b * KKY + k) * DD + t] * ysh[k];
  wsum[b * DD + t] = s;
}

__global__ void concat2_kernel(const half_t* __restrict__ agg, const float* __restrict__ hq,
                               float* __restrict__ cc2) {
  const int b = blockIdx.x, t = threadIdx.x;  // 512 threads
  cc2[b * 512 + t] = (t < HH) ? (float)agg[b * HH + t] : hq[b * DD + (t - HH)];
}

__global__ void rowdot_kernel(const half_t* __restrict__ H, int n,
                              const float* __restrict__ w, const float* __restrict__ bias,
                              float* __restrict__ out) {
  const int b = blockIdx.x, t = threadIdx.x;  // 32 threads
  float s = 0.f;
  for (int jj = t; jj < n; jj += 32) s += (float)H[(size_t)b * n + jj] * w[jj];
  for (int off = 16; off > 0; off >>= 1) s += __shfl_down(s, off, 32);
  if (t == 0) out[b] = s + bias[0];
}

__global__ void zero_terms_kernel(float* __restrict__ terms) {
  const int t = blockIdx.x * blockDim.x + threadIdx.x;
  if (t < 2 * BB) terms[BB + t] = 0.f;  // zero temporal + recency accumulators
}

__global__ void combine_kernel(const float* __restrict__ terms, const float* __restrict__ ll,
                               float* __restrict__ out) {
  const int b = threadIdx.x;  // 256
  const float l0 = log1pf(__expf(ll[0]));
  const float l1 = log1pf(__expf(ll[1]));
  const float l2 = log1pf(__expf(ll[2]));
  const float l3 = log1pf(__expf(ll[3]));
  const float l4 = log1pf(__expf(ll[4]));
  out[b] = l0 * terms[b] + l1 * terms[BB + b] + l2 * terms[2 * BB + b]
         - l3 * terms[3 * BB + b] + l4 * terms[4 * BB + b];
}

// ===========================================================================
extern "C" void kernel_launch(void* const* d_in, const int* in_sizes, int n_in,
                              void* d_out, int out_size, void* d_ws, size_t ws_size,
                              hipStream_t stream) {
  (void)in_sizes; (void)n_in; (void)out_size; (void)ws_size;
  const float* h_q  = (const float*)d_in[0];
  const float* M_   = (const float*)d_in[1];
  const float* y    = (const float*)d_in[2];
  const float* ts   = (const float*)d_in[3];
  const float* qW   = (const float*)d_in[4];
  const float* qb   = (const float*)d_in[5];
  const float* kW   = (const float*)d_in[6];
  const float* kbv  = (const float*)d_in[7];
  const float* vW   = (const float*)d_in[8];
  const float* vbv  = (const float*)d_in[9];
  const float* rh1W = (const float*)d_in[10];
  const float* rh1b = (const float*)d_in[11];
  const float* rh2W = (const float*)d_in[12];
  const float* rh2b = (const float*)d_in[13];
  const float* tw1  = (const float*)d_in[14];
  const float* tb1  = (const float*)d_in[15];
  const float* tw2  = (const float*)d_in[16];
  const float* tb2  = (const float*)d_in[17];
  const float* rw1  = (const float*)d_in[18];
  const float* rb1  = (const float*)d_in[19];
  const float* rw2  = (const float*)d_in[20];
  const float* rb2  = (const float*)d_in[21];
  const float* rw3  = (const float*)d_in[22];
  const float* rb3  = (const float*)d_in[23];
  const float* aW   = (const float*)d_in[24];
  const float* ab   = (const float*)d_in[25];
  const float* sh1W = (const float*)d_in[26];
  const float* sh1b = (const float*)d_in[27];
  const float* sh2W = (const float*)d_in[28];
  const float* sh2b = (const float*)d_in[29];
  const float* pW   = (const float*)d_in[30];
  const float* pb   = (const float*)d_in[31];
  const float* inW[2]  = {(const float*)d_in[32], (const float*)d_in[36]};
  const float* inb[2]  = {(const float*)d_in[33], (const float*)d_in[37]};
  const float* outW[2] = {(const float*)d_in[34], (const float*)d_in[38]};
  const float* outb[2] = {(const float*)d_in[35], (const float*)d_in[39]};
  const float* ch1W = (const float*)d_in[40];
  const float* ch1b = (const float*)d_in[41];
  const float* ch2W = (const float*)d_in[42];
  const float* ch2b = (const float*)d_in[43];
  const float* ll   = (const float*)d_in[44];
  float* out = (float*)d_out;

  // ---- workspace bump allocator ----
  char* base = (char*)d_ws;
  size_t cur = 0;
  auto alloc = [&](size_t bytes) -> void* {
    cur = (cur + 255) & ~(size_t)255;
    void* p = base + cur;
    cur += bytes;
    return p;
  };
  auto packHalves = [](int N, int Kd) -> size_t { return (size_t)(N / 16) * (Kd / 32) * 512; };

  half_t* P     = (half_t*)alloc((size_t)BBKK * 384 * 2);   // [Kp | V | Hm] f16
  half_t* kbuf  = (half_t*)alloc((size_t)BBKK * HH * 2);
  half_t* vbuf  = (half_t*)alloc((size_t)BBKK * HH * 2);
  half_t* BpCat = (half_t*)alloc(3 * packHalves(128, 384) * 2);
  half_t* BpRelH1  = (half_t*)alloc(packHalves(128, 512) * 2);
  half_t* BpAW     = (half_t*)alloc(packHalves(128, 384) * 2);
  half_t* BpSuffH1 = (half_t*)alloc(packHalves(128, 512) * 2);
  half_t* BpHop[2][4];
  for (int h = 0; h < 2; ++h)
    for (int m = 0; m < 4; ++m)
      BpHop[h][m] = (half_t*)alloc(packHalves(128, 128) * 2);
  half_t* BpCompH1 = (half_t*)alloc(packHalves(64, 128) * 2);
  float* biasCat = (float*)alloc(384 * 4);
  float* cc      = (float*)alloc((size_t)BB * 512 * 4);
  float* cc2     = (float*)alloc((size_t)BB * 512 * 4);
  float* wsum    = (float*)alloc((size_t)BB * DD * 4);
  half_t* stateB = (half_t*)alloc((size_t)BB * HH * 2);
  half_t* qbuf   = (half_t*)alloc((size_t)BB * HH * 2);
  half_t* ctxc   = (half_t*)alloc((size_t)BB * HH * 2);
  half_t* aggB   = (half_t*)alloc((size_t)BB * HH * 2);
  half_t* hRel   = (half_t*)alloc((size_t)BB * HH * 2);
  half_t* hSuff  = (half_t*)alloc((size_t)BB * HH * 2);
  half_t* hComp  = (half_t*)alloc((size_t)BB * 64 * 2);
  int* idx10 = (int*)alloc((size_t)BB * 10 * 4);
  int* idx8  = (int*)alloc((size_t)BB * 8 * 4);
  float* terms = (float*)alloc(5 * BB * 4);

  auto packLaunch = [&](const float* W, int N, int Kd, half_t* dst) {
    int nh = (int)packHalves(N, Kd);
    pack_b_kernel<<<(nh + 255) / 256, 256, 0, stream>>>(W, N, Kd, dst);
  };

  // ---- 0: zero accumulators, pack weights, bias concat ----
  zero_terms_kernel<<<2, 256, 0, stream>>>(terms);
  const size_t reg = packHalves(128, 384);
  packLaunch(kW, 128, 384, BpCat);
  packLaunch(vW, 128, 384, BpCat + reg);
  packLaunch(pW, 128, 384, BpCat + 2 * reg);
  packLaunch(rh1W, 128, 512, BpRelH1);
  packLaunch(aW, 128, 384, BpAW);
  packLaunch(sh1W, 128, 512, BpSuffH1);
  for (int h = 0; h < 2; ++h) {
    packLaunch(inW[h],             128, 128, BpHop[h][0]);  // q rows 0..127
    packLaunch(inW[h] + 128 * 128, 128, 128, BpHop[h][1]);  // k rows 128..255
    packLaunch(inW[h] + 256 * 128, 128, 128, BpHop[h][2]);  // v rows 256..383
    packLaunch(outW[h],            128, 128, BpHop[h][3]);
  }
  packLaunch(ch1W, 64, 128, BpCompH1);
  (void)hipMemcpyAsync(biasCat,       kbv, 128 * 4, hipMemcpyDeviceToDevice, stream);
  (void)hipMemcpyAsync(biasCat + 128, vbv, 128 * 4, hipMemcpyDeviceToDevice, stream);
  (void)hipMemcpyAsync(biasCat + 256, pb,  128 * 4, hipMemcpyDeviceToDevice, stream);

  // ---- 1: fused projection P = M @ [kW;vW;pW]^T + bias (TDM + WMMA) ----
  wmma_gemm_kernel<true, 384, 0><<<BBKK / 16, 32, 0, stream>>>(
      M_, 384, BBKK, BpCat, biasCat, P, 384);

  // ---- 2: relevance ----
  rel_attn_kernel<<<BB, 128, 0, stream>>>(h_q, qW, qb, P, y, cc);
  wmma_gemm_kernel<true, 512, 1><<<BB / 16, 32, 0, stream>>>(
      cc, 512, BB, BpRelH1, rh1b, hRel, 128);
  rowdot_kernel<<<BB, 32, 0, stream>>>(hRel, 128, rh2W, rh2b, terms);

  // ---- 3: temporal + recency ----
  topk_kernel<<<BB, 32, 0, stream>>>(y, idx10, 10);
  topk_kernel<<<BB, 32, 0, stream>>>(y, idx8, 8);
  temporal_kernel<<<BB * 45, 64, 0, stream>>>(M_, y, ts, idx10, tw1, tb1, tw2, tb2, terms + BB);
  recency_kernel<<<BB * 28, 64, 0, stream>>>(M_, y, ts, idx8, rw1, rb1, rw2, rb2, rw3, rb3,
                                             terms + 2 * BB);

  // ---- 4: sufficiency ----
  weighted_kernel<<<BB, 384, 0, stream>>>(M_, y, wsum);
  wmma_gemm_kernel<true, 384, 1><<<BB / 16, 32, 0, stream>>>(
      wsum, 384, BB, BpAW, ab, aggB, 128);
  concat2_kernel<<<BB, 512, 0, stream>>>(aggB, h_q, cc2);
  wmma_gemm_kernel<true, 512, 1><<<BB / 16, 32, 0, stream>>>(
      cc2, 512, BB, BpSuffH1, sh1b, hSuff, 128);
  rowdot_kernel<<<BB, 32, 0, stream>>>(hSuff, 128, sh2W, sh2b, terms + 3 * BB);

  // ---- 5: composition (2 MHA hops over Hm = P[:, 256:384]) ----
  wmma_gemm_kernel<true, 384, 0><<<BB / 16, 32, 0, stream>>>(
      h_q, 384, BB, BpCat + 2 * reg, biasCat + 256, stateB, 128);
  const half_t* Hm = P + 256;  // lda = 384
  for (int h = 0; h < 2; ++h) {
    wmma_gemm_kernel<false, 128, 0><<<BB / 16, 32, 0, stream>>>(
        stateB, 128, BB, BpHop[h][0], inb[h], qbuf, 128);
    wmma_gemm_kernel<false, 128, 0><<<BBKK / 16, 32, 0, stream>>>(
        Hm, 384, BBKK, BpHop[h][1], inb[h] + 128, kbuf, 128);
    wmma_gemm_kernel<false, 128, 0><<<BBKK / 16, 32, 0, stream>>>(
        Hm, 384, BBKK, BpHop[h][2], inb[h] + 256, vbuf, 128);
    mha_attn_kernel<<<BB, 128, 0, stream>>>(qbuf, kbuf, vbuf, y, ctxc);
    wmma_gemm_kernel<false, 128, 0><<<BB / 16, 32, 0, stream>>>(
        ctxc, 128, BB, BpHop[h][3], outb[h], stateB, 128);
  }
  wmma_gemm_kernel<false, 128, 1><<<BB / 16, 32, 0, stream>>>(
      stateB, 128, BB, BpCompH1, ch1b, hComp, 64);
  rowdot_kernel<<<BB, 32, 0, stream>>>(hComp, 64, ch2W, ch2b, terms + 4 * BB);

  // ---- 6: combine with softplus(log_lambdas) ----
  combine_kernel<<<1, BB, 0, stream>>>(terms, ll, out);
}
